// Tmp_attention2_70841190580533
// MI455X (gfx1250) — compile-verified, matrix operand order checked by reading, not run on hardware
//
#include <hip/hip_runtime.h>
#include <hip/hip_bf16.h>

// Problem constants (from reference): B=2, N=1024, CS=1024, CH=64, H=16, HC=1024
#define Bb   2
#define Nn   1024
#define CSc  1024
#define CHc  64
#define Hh   16
#define HCc  1024
#define INFV 100000.0f
// 2 * (1/(3*CH))^0.5 = 2/sqrt(192)
#define SSCALE 0.14433756729740643f

typedef __attribute__((ext_vector_type(16))) __bf16       v16bf;
typedef __attribute__((ext_vector_type(8)))  float        v8f;
typedef __attribute__((ext_vector_type(4)))  unsigned int u32x4;

union Frag16x32 { v16bf v; u32x4 q[2]; };

// fp32 -> bf16 round-to-nearest-even (bit-exact, no __bf16 cvt dependence)
__device__ __forceinline__ __bf16 f2bf(float x) {
  unsigned u = __float_as_uint(x);
  unsigned r = (u + 0x7FFFu + ((u >> 16) & 1u)) >> 16;
  unsigned short us = (unsigned short)r;
  __bf16 b;
  __builtin_memcpy(&b, &us, 2);
  return b;
}

__device__ __forceinline__ v8f wmma_bf16(v16bf a, v16bf b, v8f c) {
  // D = A(16x32 bf16) * B(32x16 bf16) + C(16x16 f32)
  return __builtin_amdgcn_wmma_f32_16x16x32_bf16(
      /*neg_a=*/false, a, /*neg_b=*/false, b,
      /*c_mod=*/(short)0, c, /*reuse_a=*/false, /*reuse_b=*/false);
}

// Fragment load from a per-lane row base + compile-time element offset.
// Offsets fold into the 24-bit immediate of global_load_b128.
__device__ __forceinline__ v16bf ldfrag(const __bf16* p, int eoff) {
  Frag16x32 f;
  f.q[0] = *(const u32x4*)(p + eoff);
  f.q[1] = *(const u32x4*)(p + eoff + 16);
  return f.v;
}

// General fragment load (used outside hot loops).
__device__ __forceinline__ v16bf load_frag16x32(const __bf16* base, int row0,
                                                int k0, int ld) {
  int lane = threadIdx.x & 31;
  int half = lane >> 4, lr = lane & 15;
  return ldfrag(base + (size_t)(row0 + lr) * ld + half * 8, k0);
}

// Double-buffered 32(M) x 64(N) x KDIM tile; 8 WMMAs per 12-load K-step.
// Only two live address registers; all fragment offsets are immediates.
template <int KDIM, int LDA, int LDB>
__device__ __forceinline__ void gemm_tile_32x64(const __bf16* __restrict__ A,
                                                int m0,
                                                const __bf16* __restrict__ Bt,
                                                int n0, v8f acc[8]) {
  int lane = threadIdx.x & 31;
  int half = lane >> 4, lr = lane & 15;
  const __bf16* pa = A + (size_t)(m0 + lr) * LDA + half * 8;
  const __bf16* pb = Bt + (size_t)(n0 + lr) * LDB + half * 8;

#pragma unroll
  for (int t = 0; t < 8; ++t)
#pragma unroll
    for (int i = 0; i < 8; ++i) acc[t][i] = 0.0f;

  v16bf a0 = ldfrag(pa, 0);
  v16bf a1 = ldfrag(pa, 16 * LDA);
  v16bf b0 = ldfrag(pb, 0);
  v16bf b1 = ldfrag(pb, 16 * LDB);
  v16bf b2 = ldfrag(pb, 32 * LDB);
  v16bf b3 = ldfrag(pb, 48 * LDB);

#pragma unroll 2
  for (int k = 0; k < KDIM - 32; k += 32) {
    pa += 32;
    pb += 32;
    __builtin_prefetch(pa + 32, 0, 1);
    // next-iteration loads issue before this iteration's WMMAs
    v16bf a0n = ldfrag(pa, 0);
    v16bf a1n = ldfrag(pa, 16 * LDA);
    v16bf b0n = ldfrag(pb, 0);
    v16bf b1n = ldfrag(pb, 16 * LDB);
    v16bf b2n = ldfrag(pb, 32 * LDB);
    v16bf b3n = ldfrag(pb, 48 * LDB);
    acc[0] = wmma_bf16(a0, b0, acc[0]);
    acc[1] = wmma_bf16(a0, b1, acc[1]);
    acc[2] = wmma_bf16(a0, b2, acc[2]);
    acc[3] = wmma_bf16(a0, b3, acc[3]);
    acc[4] = wmma_bf16(a1, b0, acc[4]);
    acc[5] = wmma_bf16(a1, b1, acc[5]);
    acc[6] = wmma_bf16(a1, b2, acc[6]);
    acc[7] = wmma_bf16(a1, b3, acc[7]);
    a0 = a0n; a1 = a1n;
    b0 = b0n; b1 = b1n; b2 = b2n; b3 = b3n;
  }
  // tail: last K-step, no more loads
  acc[0] = wmma_bf16(a0, b0, acc[0]);
  acc[1] = wmma_bf16(a0, b1, acc[1]);
  acc[2] = wmma_bf16(a0, b2, acc[2]);
  acc[3] = wmma_bf16(a0, b3, acc[3]);
  acc[4] = wmma_bf16(a1, b0, acc[4]);
  acc[5] = wmma_bf16(a1, b1, acc[5]);
  acc[6] = wmma_bf16(a1, b2, acc[6]);
  acc[7] = wmma_bf16(a1, b3, acc[7]);
}

// ---------------------------------------------------------------- prep kernels
__global__ void k_prep_s(const float* __restrict__ s, __bf16* __restrict__ sB,
                         int n) {
  int i = blockIdx.x * blockDim.x + threadIdx.x;
  if (i < n) sB[i] = f2bf(s[i]);
}

// Build WqkvT (3072 x 1024, row j = column j of [Wq|Wkv]) and WoT (1024 x 1024)
__global__ void k_prep_w(const float* __restrict__ Wq,
                         const float* __restrict__ Wkv,
                         const float* __restrict__ Wo,
                         __bf16* __restrict__ WqkvT,
                         __bf16* __restrict__ WoT) {
  int i = blockIdx.x * blockDim.x + threadIdx.x;
  const int n_qkv = 3072 * 1024;
  if (i < n_qkv) {
    int j = i >> 10, k = i & 1023;
    float w = (j < HCc) ? Wq[(size_t)k * HCc + j]
                        : Wkv[(size_t)k * (2 * HCc) + (j - HCc)];
    WqkvT[i] = f2bf(w);
  } else {
    int ii = i - n_qkv;
    if (ii < 1024 * 1024) {
      int nn = ii >> 10, k = ii & 1023;
      WoT[ii] = f2bf(Wo[(size_t)k * CSc + nn]);
    }
  }
}

// ------------------------------------------------------------- fused QKV GEMM
__global__ void k_gemm_qkv(const __bf16* __restrict__ sB,
                           const __bf16* __restrict__ WqkvT,
                           const float* __restrict__ bq,
                           const float* __restrict__ bkv,
                           __bf16* __restrict__ Q, __bf16* __restrict__ Kb,
                           __bf16* __restrict__ VT) {
  int wave = threadIdx.x >> 5;
  int lane = threadIdx.x & 31;
  int half = lane >> 4, lr = lane & 15;
  int m0 = blockIdx.x * 32;
  int n0 = blockIdx.y * 512 + wave * 64;

  v8f acc[8];
  gemm_tile_32x64<CSc, CSc, CSc>(sB, m0, WqkvT, n0, acc);

#pragma unroll
  for (int mt = 0; mt < 2; ++mt)
#pragma unroll
    for (int t = 0; t < 4; ++t) {
      int j = n0 + t * 16 + lr;  // output column in [0, 3072)
      float bias = (j < HCc) ? bq[j] : bkv[j - HCc];
#pragma unroll
      for (int r = 0; r < 8; ++r) {
        int m = m0 + mt * 16 + r + 8 * half;  // global row in [0, 2048)
        int bb = m >> 10, ns = m & 1023;
        __bf16 val = f2bf(acc[mt * 4 + t][r] + bias);
        if (j < HCc) {
          int h = j >> 6, c = j & 63;
          Q[(((size_t)bb * Hh + h) * Nn + ns) * CHc + c] = val;
        } else {
          int jj = j - HCc;
          int h = jj >> 7, cc = jj & 127;
          if (cc < CHc)
            Kb[(((size_t)bb * Hh + h) * Nn + ns) * CHc + cc] = val;
          else
            VT[(((size_t)bb * Hh + h) * CHc + (cc - CHc)) * Nn + ns] = val;
        }
      }
    }
}

// -------------------------------------------------- flash attention per (b,h)
__global__ void k_attn(const __bf16* __restrict__ Q,
                       const __bf16* __restrict__ Kb,
                       const __bf16* __restrict__ VT,
                       const float* __restrict__ mask,
                       __bf16* __restrict__ O) {
  extern __shared__ __bf16 lds[];  // 4 waves * 16*32 bf16 = 4 KB
  int wave = threadIdx.x >> 5;
  int lane = threadIdx.x & 31;
  int half = lane >> 4, lr = lane & 15;

  int qblk = blockIdx.x % (Nn / 64);
  int bh = blockIdx.x / (Nn / 64);
  int b = bh / Hh, h = bh % Hh;
  int q0 = qblk * 64 + wave * 16;

  const __bf16* Qbh = Q + (size_t)bh * Nn * CHc;
  const __bf16* Kbh = Kb + (size_t)bh * Nn * CHc;
  const __bf16* Vbh = VT + (size_t)bh * CHc * Nn;
  __bf16* myLds = lds + wave * (16 * 32);

  // q fragments stay resident (CH=64 -> two K=32 fragments)
  v16bf aq0 = load_frag16x32(Qbh, q0, 0, CHc);
  v16bf aq1 = load_frag16x32(Qbh, q0, 32, CHc);

  // per-lane row bases for streaming K and V; advance by 32 k-rows / k-cols
  const __bf16* pk = Kbh + (size_t)lr * CHc + half * 8;
  const __bf16* pv = Vbh + (size_t)lr * Nn + half * 8;

  float mq[8], rowmax[8], rowsum[8];
#pragma unroll
  for (int r = 0; r < 8; ++r) {
    mq[r] = mask[(size_t)b * Nn + q0 + r + 8 * half];
    rowmax[r] = -1e30f;
    rowsum[r] = 0.0f;
  }
  v8f acc[4];
#pragma unroll
  for (int t = 0; t < 4; ++t)
#pragma unroll
    for (int i = 0; i < 8; ++i) acc[t][i] = 0.0f;

  for (int kk = 0; kk < Nn; kk += 32) {
    __builtin_prefetch(pk + 32 * CHc, 0, 1);
    // ---- scores for a 16(q) x 32(k) chunk: two 16x16 tiles
    v8f sc[2];
#pragma unroll
    for (int t = 0; t < 2; ++t) {
      v16bf bk0 = ldfrag(pk, t * 16 * CHc);
      v16bf bk1 = ldfrag(pk, t * 16 * CHc + 32);
      v8f c;
#pragma unroll
      for (int i = 0; i < 8; ++i) c[i] = 0.0f;
      c = wmma_bf16(aq0, bk0, c);
      c = wmma_bf16(aq1, bk1, c);
      sc[t] = c;
    }
    pk += 32 * CHc;
    // ---- scale + mask bias
#pragma unroll
    for (int t = 0; t < 2; ++t) {
      float mk = mask[(size_t)b * Nn + kk + t * 16 + lr];
#pragma unroll
      for (int r = 0; r < 8; ++r)
        sc[t][r] = sc[t][r] * SSCALE + INFV * (mq[r] * mk - 1.0f);
    }
    // ---- online softmax update (row = r + 8*half, spread over 16 lanes)
    float corr[8];
#pragma unroll
    for (int r = 0; r < 8; ++r) {
      float m = fmaxf(sc[0][r], sc[1][r]);
      for (int d = 1; d < 16; d <<= 1) m = fmaxf(m, __shfl_xor(m, d, 32));
      float nm = fmaxf(rowmax[r], m);
      corr[r] = __expf(rowmax[r] - nm);
      rowmax[r] = nm;
      sc[0][r] = __expf(sc[0][r] - nm);
      sc[1][r] = __expf(sc[1][r] - nm);
      float s = sc[0][r] + sc[1][r];
      for (int d = 1; d < 16; d <<= 1) s += __shfl_xor(s, d, 32);
      rowsum[r] = rowsum[r] * corr[r] + s;
    }
#pragma unroll
    for (int t = 0; t < 4; ++t)
#pragma unroll
      for (int r = 0; r < 8; ++r) acc[t][r] *= corr[r];

    // ---- relayout probabilities (C layout -> A layout) through LDS
    __syncthreads();
#pragma unroll
    for (int t = 0; t < 2; ++t)
#pragma unroll
      for (int r = 0; r < 8; ++r)
        myLds[(r + 8 * half) * 32 + t * 16 + lr] = f2bf(sc[t][r]);
    __syncthreads();
    Frag16x32 ap;
    {
      const __bf16* p = myLds + lr * 32 + half * 8;
      ap.q[0] = *(const u32x4*)(p);
      ap.q[1] = *(const u32x4*)(p + 16);
    }
    // ---- o += a @ v   (B columns are VT rows -> contiguous; t offset folds)
#pragma unroll
    for (int t = 0; t < 4; ++t) {
      v16bf bv = ldfrag(pv, t * 16 * Nn);
      acc[t] = wmma_bf16(ap.v, bv, acc[t]);
    }
    pv += 32;
  }

  // ---- normalize and store O(b, n, h*CH + c) in bf16
#pragma unroll
  for (int t = 0; t < 4; ++t)
#pragma unroll
    for (int r = 0; r < 8; ++r) {
      float o = acc[t][r] / rowsum[r];
      int n = q0 + r + 8 * half;
      int c = h * CHc + t * 16 + lr;
      O[((size_t)b * Nn + n) * HCc + c] = f2bf(o);
    }
}

// --------------------------------------------------------------- output GEMM
__global__ void k_gemm_out(const __bf16* __restrict__ O,
                           const __bf16* __restrict__ WoT,
                           const float* __restrict__ bo,
                           float* __restrict__ out) {
  int wave = threadIdx.x >> 5;
  int lane = threadIdx.x & 31;
  int half = lane >> 4, lr = lane & 15;
  int m0 = blockIdx.x * 32;
  int n0 = blockIdx.y * 512 + wave * 64;

  v8f acc[8];
  gemm_tile_32x64<HCc, HCc, CSc>(O, m0, WoT, n0, acc);

#pragma unroll
  for (int mt = 0; mt < 2; ++mt)
#pragma unroll
    for (int t = 0; t < 4; ++t) {
      int j = n0 + t * 16 + lr;
      float bias = bo[j];
#pragma unroll
      for (int r = 0; r < 8; ++r) {
        int m = m0 + mt * 16 + r + 8 * half;
        out[(size_t)m * CSc + j] = acc[mt * 4 + t][r] + bias;
      }
    }
}

// ---------------------------------------------------------------------- launch
extern "C" void kernel_launch(void* const* d_in, const int* in_sizes, int n_in,
                              void* d_out, int out_size, void* d_ws,
                              size_t ws_size, hipStream_t stream) {
  (void)in_sizes; (void)n_in; (void)out_size; (void)ws_size;
  // dict order: s, z, r, mask, Wq, bq, Wkv, bkv, Wgq, bgq, Wgk, bgk, Wo, bo
  const float* s    = (const float*)d_in[0];
  const float* mask = (const float*)d_in[3];
  const float* Wq   = (const float*)d_in[4];
  const float* bq   = (const float*)d_in[5];
  const float* Wkv  = (const float*)d_in[6];
  const float* bkv  = (const float*)d_in[7];
  const float* Wo   = (const float*)d_in[12];
  const float* bo   = (const float*)d_in[13];

  char* ws = (char*)d_ws;
  __bf16* sB    = (__bf16*)(ws + (size_t)0);            // 4 MB
  __bf16* WqkvT = (__bf16*)(ws + ((size_t)4  << 20));   // 6 MB
  __bf16* WoT   = (__bf16*)(ws + ((size_t)10 << 20));   // 2 MB
  __bf16* Q     = (__bf16*)(ws + ((size_t)12 << 20));   // 4 MB
  __bf16* Kb    = (__bf16*)(ws + ((size_t)16 << 20));   // 4 MB
  __bf16* VT    = (__bf16*)(ws + ((size_t)20 << 20));   // 4 MB
  __bf16* Ob    = (__bf16*)(ws + ((size_t)24 << 20));   // 4 MB

  const int n_s = Bb * Nn * CSc;  // 2M
  k_prep_s<<<(n_s + 255) / 256, 256, 0, stream>>>(s, sB, n_s);

  const int n_w = 3072 * 1024 + 1024 * 1024;  // 4M
  k_prep_w<<<(n_w + 255) / 256, 256, 0, stream>>>(Wq, Wkv, Wo, WqkvT, WoT);

  dim3 g_qkv(64, 6);  // M/32 x 3072/512
  k_gemm_qkv<<<g_qkv, 256, 0, stream>>>(sB, WqkvT, bq, bkv, Q, Kb, VT);

  // B*H*(N/64) workgroups, 4 waves each, 4 KB dynamic LDS
  k_attn<<<Bb * Hh * (Nn / 64), 128, 4 * 16 * 32 * sizeof(__bf16), stream>>>(
      Q, Kb, VT, mask, Ob);

  dim3 g_out(64, 2);  // M/32 x 1024/512
  k_gemm_out<<<g_out, 256, 0, stream>>>(Ob, WoT, bo, (float*)d_out);
}
